// SynSAN_6975026889315
// MI455X (gfx1250) — compile-verified, matrix operand order checked by reading, not run on hardware
//
#include <hip/hip_runtime.h>
#include <hip/hip_bf16.h>
#include <math.h>

// ---------------- problem constants ----------------
#define Bn   64
#define Nn   128
#define Kn   16
#define INn  300
#define INp  320          // padded to multiple of 32
#define Hn   256
#define Cn   5
#define NEGV (-1000000000.0f)

#define ROWS_H (Bn*Nn*(1+Kn))   // 139264 rows of h
#define BNn    (Bn*Nn)          // 8192
#define KTMAX  (INp/32)         // 10 k-tiles max

typedef __attribute__((ext_vector_type(16))) __bf16 v16bf;
typedef __attribute__((ext_vector_type(8)))  float  v8f;
typedef unsigned int u32x4 __attribute__((ext_vector_type(4)));
typedef int i32x4 __attribute__((ext_vector_type(4)));
typedef int i32x8 __attribute__((ext_vector_type(8)));

#if __has_builtin(__builtin_amdgcn_tensor_load_to_lds) && __has_builtin(__builtin_amdgcn_s_wait_tensorcnt)
#define USE_TDM 1
#else
#define USE_TDM 0
#endif

// ---------------- workspace layout (bytes) ----------------
static constexpr size_t PK_WX  = 0;                                   // 320*256 bf16
static constexpr size_t PK_WH  = PK_WX + (size_t)INp*Hn*2;            // 256*256 bf16 each
static constexpr size_t PK_UH  = PK_WH + (size_t)Hn*Hn*2;
static constexpr size_t PK_WG  = PK_UH + (size_t)Hn*Hn*2;
static constexpr size_t PK_UG  = PK_WG + (size_t)Hn*Hn*2;
static constexpr size_t PK_WA  = PK_UG + (size_t)Hn*Hn*2;
static constexpr size_t PK_V   = PK_WA + (size_t)Hn*Hn*2;
static constexpr size_t H_OFF   = 1ull<<20;                           // 1 MiB aligned
static constexpr size_t H_BYTES = (size_t)ROWS_H*Hn*4;                // 142.6 MB
static constexpr size_t CWH_OFF = H_OFF   + H_BYTES;
static constexpr size_t SMALL   = (size_t)BNn*Hn*4;                   // 8.39 MB
static constexpr size_t CON_OFF = CWH_OFF + SMALL;
static constexpr size_t G1_OFF  = CON_OFF + SMALL;
static constexpr size_t G2_OFF  = G1_OFF  + SMALL;
static constexpr size_t Z_OFF   = G2_OFF  + SMALL;
static constexpr size_t S_OFF   = Z_OFF   + SMALL;
static constexpr size_t AVL_OFF = S_OFF   + SMALL;
static constexpr size_t SENT_OFF= AVL_OFF + SMALL;                    // 64*256 f32

// K position inside a 16-bit A fragment: pair index v (0..7), lane-half h
__device__ __forceinline__ int frag_kbase(int v, int half) {
    return (v < 4) ? (2*v + 8*half) : (16 + 2*(v-4) + 8*half);
}

__device__ __forceinline__ v8f wmma_bf16(v16bf a, v16bf b, v8f c) {
    return __builtin_amdgcn_wmma_f32_16x16x32_bf16(false, a, false, b, (short)0, c,
                                                   false, false);
}

#if USE_TDM
// ---- Tensor Data Mover: 2D tile (rows x tile_d0 elems, f32) -> LDS ---------
// tensor_dim0 = valid row length (OOB columns return zero => free K padding),
// tensor_dim0_stride = row pitch in elements (supports strided "center" rows).
__device__ __forceinline__ void tdm_load_2d(unsigned lds_addr, const float* gptr,
                                            unsigned tensor_d0, unsigned tile_d0,
                                            unsigned rows, unsigned long long stride0) {
    unsigned long long ga = (unsigned long long)(uintptr_t)gptr;
    u32x4 g0;
    g0[0] = 1u;                                               // count=1, user D#
    g0[1] = lds_addr;                                         // LDS byte address
    g0[2] = (unsigned)ga;                                     // global_addr[31:0]
    g0[3] = (unsigned)((ga >> 32) & 0x01FFFFFFu) | (2u << 30);// ga[56:32] | type=2
    i32x8 g1;
    g1[0] = (int)(2u << 16);                                  // data_size=4B
    g1[1] = (int)((tensor_d0 & 0xFFFFu) << 16);               // tensor_dim0 lo
    g1[2] = (int)(((tensor_d0 >> 16) & 0xFFFFu) | ((rows & 0xFFFFu) << 16)); // td0 hi | tensor_dim1
    g1[3] = (int)((tile_d0 & 0xFFFFu) << 16);                 // td1 hi=0 | tile_dim0
    g1[4] = (int)(rows & 0xFFFFu);                            // tile_dim1, tile_dim2=0
    g1[5] = (int)(unsigned)(stride0 & 0xFFFFFFFFull);         // dim0 stride lo
    g1[6] = (int)(unsigned)((stride0 >> 32) & 0xFFFFull);     // dim0 stride hi
    g1[7] = 0;
    i32x4 gz = {0, 0, 0, 0};
#if defined(__clang_major__) && __clang_major__ >= 23
    i32x8 gz8 = {0, 0, 0, 0, 0, 0, 0, 0};
    __builtin_amdgcn_tensor_load_to_lds(g0, g1, gz, gz, gz8, 0);
#else
    __builtin_amdgcn_tensor_load_to_lds(g0, g1, gz, gz, 0);
#endif
}
#endif

// ---------------- weight repack: f32 (K x N row-major) -> bf16 B fragments ----
// dst index = ((nt*KT + kt)*32 + lane)*16 + e
__global__ void pack_w(const float* __restrict__ W, __bf16* __restrict__ dst,
                       int Kact, int KT, int Ncols) {
    int t = blockIdx.x * 256 + threadIdx.x;
    if (t >= KT * 32 * 16 * (Ncols/16)) return;
    int e    = t & 15;
    int lane = (t >> 4) & 31;
    int rem  = t >> 9;
    int kt   = rem % KT;
    int nt   = rem / KT;
    int half = lane >> 4;
    int k = kt*32 + frag_kbase(e >> 1, half) + (e & 1);
    int n = nt*16 + (lane & 15);
    float val = (k < Kact) ? W[(size_t)k * Ncols + n] : 0.0f;
    dst[t] = (__bf16)val;
}

// Cooperative f32 panel -> bf16 fragment-layout LDS conversion (one m-tile).
// ldsAf[(mt*KT+kt)*512 + lane*16 + 2v] ; panel row (mt*16+m), col k.
__device__ __forceinline__ void convert_panel(const float* __restrict__ ldsA,
                                              __bf16* __restrict__ ldsAf,
                                              int mt, int KT, int Kp, int tid) {
    for (int kt = 0; kt < KT; ++kt) {
        for (int p = tid; p < 256; p += 128) {
            int e2   = p & 7;                 // pair index v
            int lane = p >> 3;
            int half = lane >> 4;
            int k    = kt*32 + frag_kbase(e2, half);
            int row  = mt*16 + (lane & 15);
            float2 q = *(const float2*)(ldsA + (size_t)row * Kp + k);
            __bf16* dst = ldsAf + (((size_t)mt*KT + kt) << 9) + lane*16 + e2*2;
            dst[0] = (__bf16)q.x;
            dst[1] = (__bf16)q.y;
        }
    }
}

// ---------------- generic WMMA GEMM: C[32*gridDim.x x 256] ------------------
// A physical row for panel row i of tile tm:  tm*tm_mul + i*i_mul + row_off
// TDM stages a 32 x Kp f32 panel; one cooperative pass converts it to bf16
// fragments; 4 waves x (2 m-tiles x 4 n-tiles) = 8 WMMAs per k-step per wave.
// epi: 0 = none, 1 = tanh, 2 = sigmoid. bias added per output column.
__global__ __launch_bounds__(128)
void gemm_bf16(const float* __restrict__ A, const __bf16* __restrict__ Bpk,
               const float* __restrict__ bias, float* __restrict__ C,
               int lda, int Kact, int Kp,
               int tm_mul, int i_mul, int row_off, int epi) {
    __shared__ float  ldsA[32 * INp];                   // 40 KB raw f32 panel
    __shared__ __bf16 ldsAf[2 * KTMAX * 512];           // 20 KB bf16 fragments
    const int tid  = threadIdx.x;
    const int lane = tid & 31;
    const int w    = tid >> 5;                          // 4 waves
    const int half = lane >> 4;
    const int tm   = blockIdx.x;
    const int KT   = Kp >> 5;

#if USE_TDM
    if (tid < 32) {
        const float* gbase = A + ((size_t)tm * tm_mul + row_off) * lda;
        tdm_load_2d((unsigned)(uintptr_t)ldsA, gbase,
                    (unsigned)Kact, (unsigned)Kp, 32u,
                    (unsigned long long)((long long)i_mul * lda));
        __builtin_amdgcn_s_wait_tensorcnt(0);
    }
#else
    for (int idx = tid; idx < 32 * Kp; idx += 128) {
        int r = idx / Kp, k = idx - r * Kp;
        long prow = (long)tm * tm_mul + (long)r * i_mul + row_off;
        ldsA[idx] = (k < Kact) ? A[(size_t)prow * lda + k] : 0.0f;
    }
#endif
    __syncthreads();
    convert_panel(ldsA, ldsAf, 0, KT, Kp, tid);
    convert_panel(ldsA, ldsAf, 1, KT, Kp, tid);
    __syncthreads();

    const __bf16* b0 = Bpk + ((size_t)(4*w + 0) * KT) * 512 + lane * 16;
    const __bf16* b1 = Bpk + ((size_t)(4*w + 1) * KT) * 512 + lane * 16;
    const __bf16* b2 = Bpk + ((size_t)(4*w + 2) * KT) * 512 + lane * 16;
    const __bf16* b3 = Bpk + ((size_t)(4*w + 3) * KT) * 512 + lane * 16;
    const __bf16* af = ldsAf + lane * 16;

    v8f acc00 = {}, acc01 = {}, acc02 = {}, acc03 = {};
    v8f acc10 = {}, acc11 = {}, acc12 = {}, acc13 = {};
    for (int kt = 0; kt < KT; ++kt) {
        v16bf a0 = *(const v16bf*)(af + ((size_t)kt << 9));
        v16bf a1 = *(const v16bf*)(af + (((size_t)KT + kt) << 9));
        v16bf f0 = *(const v16bf*)(b0 + (size_t)kt * 512);
        v16bf f1 = *(const v16bf*)(b1 + (size_t)kt * 512);
        v16bf f2 = *(const v16bf*)(b2 + (size_t)kt * 512);
        v16bf f3 = *(const v16bf*)(b3 + (size_t)kt * 512);
        acc00 = wmma_bf16(a0, f0, acc00);  acc10 = wmma_bf16(a1, f0, acc10);
        acc01 = wmma_bf16(a0, f1, acc01);  acc11 = wmma_bf16(a1, f1, acc11);
        acc02 = wmma_bf16(a0, f2, acc02);  acc12 = wmma_bf16(a1, f2, acc12);
        acc03 = wmma_bf16(a0, f3, acc03);  acc13 = wmma_bf16(a1, f3, acc13);
    }

    v8f accs[2][4] = {{acc00, acc01, acc02, acc03}, {acc10, acc11, acc12, acc13}};
#pragma unroll
    for (int mt = 0; mt < 2; ++mt) {
#pragma unroll
        for (int j = 0; j < 4; ++j) {
            int nt = 4*w + j;
            int f  = nt * 16 + (lane & 15);
            float bv = bias[f];
#pragma unroll
            for (int v = 0; v < 8; ++v) {
                float val = accs[mt][j][v] + bv;
                if (epi == 1)      val = tanhf(val);
                else if (epi == 2) val = 1.0f / (1.0f + expf(-val));
                int row = tm * 32 + mt * 16 + v + 8 * half;
                C[(size_t)row * Hn + f] = val;
            }
        }
    }
}

// ---- fused neigh@Uh + sigmoid + masked softmax over 16 neighbors + con -----
// One block per (b,n). The 16 WMMA rows == the 16 neighbors; the TDM-staged
// LDS panel doubles as the neighbor-h matrix for the a*neigh reduction.
__global__ __launch_bounds__(128)
void attn_con(const float* __restrict__ h, const __bf16* __restrict__ UhPk,
              const float* __restrict__ Uh_b, const float* __restrict__ cWh,
              const int* __restrict__ cnts, float* __restrict__ con) {
    __shared__ float  ldsA[16 * Hn];                    // 16 neighbors x 256 f32
    __shared__ __bf16 ldsAf[8 * 512];                   // 8 KB bf16 fragments
    const int tid  = threadIdx.x;
    const int lane = tid & 31;
    const int w    = tid >> 5;
    const int half = lane >> 4;
    const int bn   = blockIdx.x;

#if USE_TDM
    if (tid < 32) {
        const float* gbase = h + ((size_t)bn * 17 + 1) * Hn;   // neighbor rows
        tdm_load_2d((unsigned)(uintptr_t)ldsA, gbase,
                    (unsigned)Hn, (unsigned)Hn, 16u, (unsigned long long)Hn);
        __builtin_amdgcn_s_wait_tensorcnt(0);
    }
#else
    for (int idx = tid; idx < 16 * Hn; idx += 128) {
        int r = idx >> 8, k = idx & 255;
        ldsA[idx] = h[((size_t)bn * 17 + 1 + r) * Hn + k];
    }
#endif
    __syncthreads();
    convert_panel(ldsA, ldsAf, 0, 8, Hn, tid);
    __syncthreads();

    const __bf16* af = ldsAf + lane * 16;
    const __bf16* bb[4];
#pragma unroll
    for (int j = 0; j < 4; ++j)
        bb[j] = UhPk + ((size_t)(4*w + j) * 8) * 512 + lane * 16;

    v8f acc0 = {}, acc1 = {}, acc2 = {}, acc3 = {};
    for (int kt = 0; kt < 8; ++kt) {
        v16bf a = *(const v16bf*)(af + ((size_t)kt << 9));
        acc0 = wmma_bf16(a, *(const v16bf*)(bb[0] + (size_t)kt * 512), acc0);
        acc1 = wmma_bf16(a, *(const v16bf*)(bb[1] + (size_t)kt * 512), acc1);
        acc2 = wmma_bf16(a, *(const v16bf*)(bb[2] + (size_t)kt * 512), acc2);
        acc3 = wmma_bf16(a, *(const v16bf*)(bb[3] + (size_t)kt * 512), acc3);
    }

    const int cnt = cnts[bn];
    v8f accs[4] = {acc0, acc1, acc2, acc3};
#pragma unroll
    for (int j = 0; j < 4; ++j) {
        int nt = 4*w + j;
        int f  = nt * 16 + (lane & 15);
        float add = cWh[(size_t)bn * Hn + f] + Uh_b[f];   // cWh already has Wh_b

        float t[8];
#pragma unroll
        for (int v = 0; v < 8; ++v) {
            int m = v + 8 * half;                          // neighbor index
            float sg = 1.0f / (1.0f + expf(-(accs[j][v] + add)));
            t[v] = (m < cnt) ? sg : NEGV;
        }
        float mx = t[0];
#pragma unroll
        for (int v = 1; v < 8; ++v) mx = fmaxf(mx, t[v]);
        mx = fmaxf(mx, __shfl_xor(mx, 16, 32));

        float e[8], s = 0.0f;
#pragma unroll
        for (int v = 0; v < 8; ++v) { e[v] = expf(t[v] - mx); s += e[v]; }
        s += __shfl_xor(s, 16, 32);

        float cp = 0.0f;
#pragma unroll
        for (int v = 0; v < 8; ++v) {
            int m = v + 8 * half;
            cp += (e[v] / s) * ldsA[(size_t)m * Hn + f];   // neighbor h from LDS
        }
        cp += __shfl_xor(cp, 16, 32);
        if (half == 0) con[(size_t)bn * Hn + f] = cp;
    }
}

// ---------------- gate / z elementwise ----------------
__global__ void gate_z(const float* __restrict__ g1, const float* __restrict__ g2,
                       const float* __restrict__ h,  const float* __restrict__ con,
                       float* __restrict__ z) {
    int bn = blockIdx.x, f = threadIdx.x;
    size_t i = (size_t)bn * Hn + f;
    float g  = 1.0f / (1.0f + expf(-(g1[i] + g2[i])));
    float ch = h[((size_t)bn * 17) * Hn + f];              // center row
    z[i] = g * ch + (1.0f - g) * con[i];
}

// ------- softmax over the N=128 nodes per (b, feature), then sent ----------
__global__ void node_softmax_sent(const float* __restrict__ avl,
                                  const float* __restrict__ z,
                                  float* __restrict__ sent) {
    int b = blockIdx.x, f = threadIdx.x;                   // 64 blocks x 256 thr
    const float* col = avl + ((size_t)b * Nn) * Hn + f;
    const float* zc  = z   + ((size_t)b * Nn) * Hn + f;
    float mx = -1e30f;
    for (int n = 0; n < Nn; ++n) mx = fmaxf(mx, col[(size_t)n * Hn]);
    float s = 0.0f, acc = 0.0f;
    for (int n = 0; n < Nn; ++n) {
        float ev = expf(col[(size_t)n * Hn] - mx);
        s   += ev;
        acc += ev * zc[(size_t)n * Hn];
    }
    sent[(size_t)b * Hn + f] = acc / s;
}

// ---------------- final 64x5 classifier + log_softmax ----------------------
__global__ void final_logits(const float* __restrict__ sent,
                             const float* __restrict__ Wc,
                             const float* __restrict__ Wc_b,
                             float* __restrict__ out) {
    int b = threadIdx.x;                                   // 1 block x 64 thr
    if (b >= Bn) return;
    float l[Cn];
#pragma unroll
    for (int c = 0; c < Cn; ++c) {
        float d = Wc_b[c];
        for (int f = 0; f < Hn; ++f) d += sent[(size_t)b * Hn + f] * Wc[(size_t)f * Cn + c];
        l[c] = d;
    }
    float mx = l[0];
#pragma unroll
    for (int c = 1; c < Cn; ++c) mx = fmaxf(mx, l[c]);
    float s = 0.0f;
#pragma unroll
    for (int c = 0; c < Cn; ++c) s += expf(l[c] - mx);
    float lse = mx + logf(s);
#pragma unroll
    for (int c = 0; c < Cn; ++c) out[(size_t)b * Cn + c] = l[c] - lse;
}

// ---------------- host launcher ----------------
extern "C" void kernel_launch(void* const* d_in, const int* in_sizes, int n_in,
                              void* d_out, int out_size, void* d_ws, size_t ws_size,
                              hipStream_t stream) {
    const float* x     = (const float*)d_in[0];
    const int*   cnts  = (const int*)  d_in[1];
    const float* Wx_w  = (const float*)d_in[2];  const float* Wx_b = (const float*)d_in[3];
    const float* Wh_w  = (const float*)d_in[4];  const float* Wh_b = (const float*)d_in[5];
    const float* Uh_w  = (const float*)d_in[6];  const float* Uh_b = (const float*)d_in[7];
    const float* Wg_w  = (const float*)d_in[8];  const float* Wg_b = (const float*)d_in[9];
    const float* Ug_w  = (const float*)d_in[10]; const float* Ug_b = (const float*)d_in[11];
    const float* Wa_w  = (const float*)d_in[12]; const float* Wa_b = (const float*)d_in[13];
    const float* V_w   = (const float*)d_in[14]; const float* V_b  = (const float*)d_in[15];
    const float* Wc_w  = (const float*)d_in[16]; const float* Wc_b = (const float*)d_in[17];
    float* out = (float*)d_out;

    char* ws = (char*)d_ws;
    __bf16* pWx = (__bf16*)(ws + PK_WX);
    __bf16* pWh = (__bf16*)(ws + PK_WH);
    __bf16* pUh = (__bf16*)(ws + PK_UH);
    __bf16* pWg = (__bf16*)(ws + PK_WG);
    __bf16* pUg = (__bf16*)(ws + PK_UG);
    __bf16* pWa = (__bf16*)(ws + PK_WA);
    __bf16* pV  = (__bf16*)(ws + PK_V);
    float* h    = (float*)(ws + H_OFF);
    float* cWh  = (float*)(ws + CWH_OFF);
    float* con  = (float*)(ws + CON_OFF);
    float* g1   = (float*)(ws + G1_OFF);
    float* g2   = (float*)(ws + G2_OFF);
    float* z    = (float*)(ws + Z_OFF);
    float* sbuf = (float*)(ws + S_OFF);
    float* avl  = (float*)(ws + AVL_OFF);
    float* sent = (float*)(ws + SENT_OFF);

    // 1) repack weights to bf16 WMMA B fragments
    pack_w<<<(INp*Hn)/256, 256, 0, stream>>>(Wx_w, pWx, INn, INp/32, Hn);
    pack_w<<<(Hn*Hn)/256,  256, 0, stream>>>(Wh_w, pWh, Hn, Hn/32, Hn);
    pack_w<<<(Hn*Hn)/256,  256, 0, stream>>>(Uh_w, pUh, Hn, Hn/32, Hn);
    pack_w<<<(Hn*Hn)/256,  256, 0, stream>>>(Wg_w, pWg, Hn, Hn/32, Hn);
    pack_w<<<(Hn*Hn)/256,  256, 0, stream>>>(Ug_w, pUg, Hn, Hn/32, Hn);
    pack_w<<<(Hn*Hn)/256,  256, 0, stream>>>(Wa_w, pWa, Hn, Hn/32, Hn);
    pack_w<<<(Hn*Hn)/256,  256, 0, stream>>>(V_w,  pV,  Hn, Hn/32, Hn);

    // 2) h = tanh(x @ Wx + b)   (139264 x 300) @ (300 x 256), K zero-padded by TDM
    gemm_bf16<<<ROWS_H/32, 128, 0, stream>>>(x, pWx, Wx_b, h,
                                             INn, INn, INp, 32, 1, 0, 1);
    // 3) cWh = center @ Wh + Wh_b   (center rows: stride 17 in h, via TDM pitch)
    gemm_bf16<<<BNn/32, 128, 0, stream>>>(h, pWh, Wh_b, cWh,
                                          Hn, Hn, Hn, 544, 17, 0, 0);
    // 4) fused neigh@Uh + sigmoid + masked softmax over 16 neighbors -> con
    attn_con<<<BNn, 128, 0, stream>>>(h, pUh, Uh_b, cWh, cnts, con);
    // 5) gate inputs
    gemm_bf16<<<BNn/32, 128, 0, stream>>>(h,   pWg, Wg_b, g1,
                                          Hn, Hn, Hn, 544, 17, 0, 0);
    gemm_bf16<<<BNn/32, 128, 0, stream>>>(con, pUg, Ug_b, g2,
                                          Hn, Hn, Hn, 32, 1, 0, 0);
    gate_z<<<BNn, Hn, 0, stream>>>(g1, g2, h, con, z);
    // 6) node attention
    gemm_bf16<<<BNn/32, 128, 0, stream>>>(z,    pWa, Wa_b, sbuf,
                                          Hn, Hn, Hn, 32, 1, 0, 2);
    gemm_bf16<<<BNn/32, 128, 0, stream>>>(sbuf, pV,  V_b,  avl,
                                          Hn, Hn, Hn, 32, 1, 0, 0);
    node_softmax_sent<<<Bn, Hn, 0, stream>>>(avl, z, sent);
    // 7) classifier + log_softmax
    final_logits<<<1, 64, 0, stream>>>(sent, Wc_w, Wc_b, out);
}
